// Graph_PhysNet_59614146068750
// MI455X (gfx1250) — compile-verified, matrix operand order checked by reading, not run on hardware
//
#include <hip/hip_runtime.h>

#define F 128
#define R 64
#define NBLK 5

typedef __attribute__((ext_vector_type(16))) __bf16 v16bf;
typedef __attribute__((ext_vector_type(8)))  float  v8f;

enum { MODE_BIAS_ACT = 0, MODE_BIAS_RESID = 1, MODE_GATED = 2 };

// shifted softplus: log(exp(x)+1) - log(2), numerically stable
__device__ __forceinline__ float ssp(float v) {
    return fmaxf(v, 0.f) + log1pf(__expf(-fabsf(v))) - 0.6931471805599453f;
}

// float -> bf16 round-to-nearest-even (storage only; WMMA consumes it)
__device__ __forceinline__ __bf16 to_bf16(float f) {
    union { float f; unsigned u; } in; in.f = f;
    unsigned r = (in.u + 0x7FFFu + ((in.u >> 16) & 1u)) >> 16;
    union { unsigned short s; __bf16 b; } out; out.s = (unsigned short)r;
    return out.b;
}

// ---------------- elementwise kernels ----------------

__global__ void act_bf16_kernel(const float* __restrict__ src,
                                __bf16* __restrict__ dst, long n) {
    long i = (long)blockIdx.x * blockDim.x + threadIdx.x;
    if (i < n) dst[i] = to_bf16(ssp(src[i]));
}

__global__ void copy_f32_kernel(const float* __restrict__ src,
                                float* __restrict__ dst, long n) {
    long i = (long)blockIdx.x * blockDim.x + threadIdx.x;
    if (i < n) dst[i] = src[i];
}

// descriptors = cutoffs * rbfs, stored bf16 (P x R)
__global__ void desc_kernel(const float* __restrict__ cutoffs,
                            const float* __restrict__ rbfs,
                            __bf16* __restrict__ dst, long total) {
    long i = (long)blockIdx.x * blockDim.x + threadIdx.x;
    if (i < total) {
        long p = i >> 6;  // R == 64
        dst[i] = to_bf16(cutoffs[p] * rbfs[i]);
    }
}

// W (K x Ncol fp32, row-major) -> Wt (Ncol x K bf16) so B fragments load
// contiguously per output column
__global__ void transpose_w_kernel(const float* __restrict__ src,
                                   __bf16* __restrict__ dst, int K, int Ncol) {
    int i = blockIdx.x * blockDim.x + threadIdx.x;
    if (i < K * Ncol) {
        int k = i / Ncol, n = i % Ncol;
        dst[(long)n * K + k] = to_bf16(src[i]);
    }
}

// ---------------- WMMA GEMM: (N x 128) @ (128 x 128) ----------------
// 256 threads = 8 waves; each wave computes a 16-row x 128-col tile with
// 8 v_wmma_f32_16x16x32_bf16 accum tiles, K loop of 4 x 32.
// A-frag per lane (h = lane>>4, m = lane&15): row m, K in
// [K0+8h, K0+8h+8) and [K0+16+8h, K0+16+8h+8)  -> two b128 loads.
// B-frag identical pattern on the pre-transposed (N-major) weight.
// C/D layout: VGPR r -> row r+8h, col = lane&15.
__global__ __launch_bounds__(256) void gemm128_kernel(
    const __bf16* __restrict__ A, const __bf16* __restrict__ Wt,
    const float* __restrict__ bias,
    float* __restrict__ out_f32, __bf16* __restrict__ out_bf16,
    float* __restrict__ out2_f32,
    const float* __restrict__ resid, const float* __restrict__ gate,
    int N, int mode)
{
    const int lane = threadIdx.x & 31;
    const int wave = threadIdx.x >> 5;
    const int h    = lane >> 4;
    const int ml   = lane & 15;
    const long row0 = ((long)blockIdx.x * 8 + wave) * 16;
    if (row0 >= N) return;

    v8f acc[8];
#pragma unroll
    for (int t = 0; t < 8; ++t) acc[t] = (v8f){0.f,0.f,0.f,0.f,0.f,0.f,0.f,0.f};

    long arow = row0 + ml; if (arow > (long)N - 1) arow = (long)N - 1;
    const uint4* aptr = reinterpret_cast<const uint4*>(A + arow * F); // 16 uint4/row

#pragma unroll
    for (int kt = 0; kt < 4; ++kt) {
        union { v16bf v; uint4 q[2]; } af;
        af.q[0] = aptr[kt * 4 + h];
        af.q[1] = aptr[kt * 4 + 2 + h];
#pragma unroll
        for (int t = 0; t < 8; ++t) {
            const uint4* bptr =
                reinterpret_cast<const uint4*>(Wt + (long)(t * 16 + ml) * F);
            union { v16bf v; uint4 q[2]; } bfr;
            bfr.q[0] = bptr[kt * 4 + h];
            bfr.q[1] = bptr[kt * 4 + 2 + h];
            acc[t] = __builtin_amdgcn_wmma_f32_16x16x32_bf16(
                false, af.v, false, bfr.v, (short)0, acc[t], false, false);
        }
    }

#pragma unroll
    for (int t = 0; t < 8; ++t) {
        const int col = t * 16 + ml;
        const float b = bias[col];
        const float g = (mode == MODE_GATED) ? gate[col] : 0.f;
#pragma unroll
        for (int r = 0; r < 8; ++r) {
            long row = row0 + r + 8 * h;
            if (row >= N) continue;
            float v = acc[t][r] + b;
            if (mode == MODE_BIAS_ACT)        v = ssp(v);
            else if (mode == MODE_BIAS_RESID) v += resid[row * F + col];
            else                              v += g * resid[row * F + col];
            if (out_f32)  out_f32 [row * F + col] = v;
            if (out_bf16) out_bf16[row * F + col] = to_bf16(v);
            if (out2_f32) out2_f32[row * F + col] = v;
        }
    }
}

// ---------------- pair kernel: g = desc @ Wg, fused gather/scatter ----------
// Wave handles 16 pairs: WMMA over K=64 (2 steps), then per element:
// atomicAdd(m[idx_i[p]][col], g * j_msg[idx_j[p]][col])
__global__ __launch_bounds__(256) void pair_kernel(
    const __bf16* __restrict__ Dsc, const __bf16* __restrict__ WgT,
    const int* __restrict__ idx_i, const int* __restrict__ idx_j,
    const float* __restrict__ jmsg, float* __restrict__ mbuf, int P)
{
    const int lane = threadIdx.x & 31;
    const int wave = threadIdx.x >> 5;
    const int h    = lane >> 4;
    const int ml   = lane & 15;
    const long p0 = ((long)blockIdx.x * 8 + wave) * 16;
    if (p0 >= P) return;

    v8f acc[8];
#pragma unroll
    for (int t = 0; t < 8; ++t) acc[t] = (v8f){0.f,0.f,0.f,0.f,0.f,0.f,0.f,0.f};

    long ar = p0 + ml; if (ar > (long)P - 1) ar = (long)P - 1;
    const uint4* aptr = reinterpret_cast<const uint4*>(Dsc + ar * R); // 8 uint4/row

#pragma unroll
    for (int kt = 0; kt < 2; ++kt) {
        union { v16bf v; uint4 q[2]; } af;
        af.q[0] = aptr[kt * 4 + h];
        af.q[1] = aptr[kt * 4 + 2 + h];
#pragma unroll
        for (int t = 0; t < 8; ++t) {
            const uint4* bptr =
                reinterpret_cast<const uint4*>(WgT + (long)(t * 16 + ml) * R);
            union { v16bf v; uint4 q[2]; } bfr;
            bfr.q[0] = bptr[kt * 4 + h];
            bfr.q[1] = bptr[kt * 4 + 2 + h];
            acc[t] = __builtin_amdgcn_wmma_f32_16x16x32_bf16(
                false, af.v, false, bfr.v, (short)0, acc[t], false, false);
        }
    }

#pragma unroll
    for (int t = 0; t < 8; ++t) {
        const int col = t * 16 + ml;
#pragma unroll
        for (int r = 0; r < 8; ++r) {
            long p = p0 + r + 8 * h;
            if (p >= P) continue;
            int ii = idx_i[p];
            int jj = idx_j[p];
            float val = acc[t][r] * jmsg[(long)jj * F + col];
            atomicAdd(&mbuf[(long)ii * F + col], val);
        }
    }
}

// ---------------- host orchestration ----------------

extern "C" void kernel_launch(void* const* d_in, const int* in_sizes, int n_in,
                              void* d_out, int out_size, void* d_ws, size_t ws_size,
                              hipStream_t stream) {
    const float* features = (const float*)d_in[0];
    // d_in[1] distances: unused by the reference
    const float* cutoffs = (const float*)d_in[2];
    const float* rbfs    = (const float*)d_in[3];
    const int*   idx_i   = (const int*)d_in[4];
    const int*   idx_j   = (const int*)d_in[5];
    const float* Wg   = (const float*)d_in[6];
    const float* Wi   = (const float*)d_in[7];
    const float* bi   = (const float*)d_in[8];
    const float* Wj   = (const float*)d_in[9];
    const float* bj   = (const float*)d_in[10];
    const float* Wri1 = (const float*)d_in[11];
    const float* bri1 = (const float*)d_in[12];
    const float* Wri2 = (const float*)d_in[13];
    const float* bri2 = (const float*)d_in[14];
    const float* Wd   = (const float*)d_in[15];
    const float* bd   = (const float*)d_in[16];
    const float* u    = (const float*)d_in[17];
    const float* Wrf1 = (const float*)d_in[18];
    const float* brf1 = (const float*)d_in[19];
    const float* Wrf2 = (const float*)d_in[20];
    const float* brf2 = (const float*)d_in[21];

    const long N  = in_sizes[0] / F;
    const long P  = in_sizes[4];
    const long NF = N * F;
    float* out = (float*)d_out;

    // ---- workspace carve-up ----
    char* w = (char*)d_ws;
    float*  x     = (float*)w;  w += NF * 4;
    float*  mbuf  = (float*)w;  w += NF * 4;
    float*  jmsg  = (float*)w;  w += NF * 4;
    __bf16* xa    = (__bf16*)w; w += NF * 2;
    __bf16* ybf   = (__bf16*)w; w += NF * 2;
    __bf16* WgT   = (__bf16*)w; w += (long)NBLK * F * R * 2;
    __bf16* WiT   = (__bf16*)w; w += (long)NBLK * F * F * 2;
    __bf16* WjT   = (__bf16*)w; w += (long)NBLK * F * F * 2;
    __bf16* Wri1T = (__bf16*)w; w += (long)NBLK * 3 * F * F * 2;
    __bf16* Wri2T = (__bf16*)w; w += (long)NBLK * 3 * F * F * 2;
    __bf16* WdT   = (__bf16*)w; w += (long)NBLK * F * F * 2;
    __bf16* Wrf1T = (__bf16*)w; w += (long)NBLK * 2 * F * F * 2;
    __bf16* Wrf2T = (__bf16*)w; w += (long)NBLK * 2 * F * F * 2;
    __bf16* desc  = (__bf16*)w; w += P * R * 2;
    (void)ws_size; (void)n_in; (void)out_size;

    const int ewGrid   = (int)((NF + 255) / 256);
    const int gemmGrid = (int)((N + 127) / 128);   // 8 waves * 16 rows / block
    const int pairGrid = (int)((P + 127) / 128);

    // ---- one-time per-call prep ----
    copy_f32_kernel<<<ewGrid, 256, 0, stream>>>(features, x, NF);
    desc_kernel<<<(int)((P * R + 255) / 256), 256, 0, stream>>>(cutoffs, rbfs, desc, P * R);

    auto TW = [&](const float* src, __bf16* dst, int K, int Ncol) {
        transpose_w_kernel<<<(K * Ncol + 255) / 256, 256, 0, stream>>>(src, dst, K, Ncol);
    };
    for (int b = 0; b < NBLK; ++b) {
        TW(Wg + (long)b * R * F, WgT + (long)b * F * R, R, F);
        TW(Wi + (long)b * F * F, WiT + (long)b * F * F, F, F);
        TW(Wj + (long)b * F * F, WjT + (long)b * F * F, F, F);
        TW(Wd + (long)b * F * F, WdT + (long)b * F * F, F, F);
        for (int r = 0; r < 3; ++r) {
            TW(Wri1 + ((long)b * 3 + r) * F * F, Wri1T + ((long)b * 3 + r) * F * F, F, F);
            TW(Wri2 + ((long)b * 3 + r) * F * F, Wri2T + ((long)b * 3 + r) * F * F, F, F);
        }
        for (int r = 0; r < 2; ++r) {
            TW(Wrf1 + ((long)b * 2 + r) * F * F, Wrf1T + ((long)b * 2 + r) * F * F, F, F);
            TW(Wrf2 + ((long)b * 2 + r) * F * F, Wrf2T + ((long)b * 2 + r) * F * F, F, F);
        }
    }

    auto gemm = [&](const __bf16* A, const __bf16* Wt, const float* bias,
                    float* of32, __bf16* obf, float* o2,
                    const float* resid, const float* gate, int mode) {
        gemm128_kernel<<<gemmGrid, 256, 0, stream>>>(A, Wt, bias, of32, obf, o2,
                                                     resid, gate, (int)N, mode);
    };

    // ---- 5 interaction blocks ----
    for (int b = 0; b < NBLK; ++b) {
        // xa = bf16(act(x))
        act_bf16_kernel<<<ewGrid, 256, 0, stream>>>(x, xa, NF);
        // i_msg -> mbuf ; j_msg -> jmsg
        gemm(xa, WiT + (long)b * F * F, bi + (long)b * F, mbuf, nullptr, nullptr,
             nullptr, nullptr, MODE_BIAS_ACT);
        gemm(xa, WjT + (long)b * F * F, bj + (long)b * F, jmsg, nullptr, nullptr,
             nullptr, nullptr, MODE_BIAS_ACT);
        // mbuf += segment_sum(g * j_msg[idx_j])
        pair_kernel<<<pairGrid, 256, 0, stream>>>(desc, WgT + (long)b * F * R,
                                                  idx_i, idx_j, jmsg, mbuf, (int)P);
        // 3 interaction residual layers on mbuf
        for (int r = 0; r < 3; ++r) {
            act_bf16_kernel<<<ewGrid, 256, 0, stream>>>(mbuf, xa, NF);
            gemm(xa, Wri1T + ((long)b * 3 + r) * F * F, bri1 + ((long)b * 3 + r) * F,
                 nullptr, ybf, nullptr, nullptr, nullptr, MODE_BIAS_ACT);
            gemm(ybf, Wri2T + ((long)b * 3 + r) * F * F, bri2 + ((long)b * 3 + r) * F,
                 mbuf, nullptr, nullptr, mbuf, nullptr, MODE_BIAS_RESID);
        }
        // x = u*x + act(m) @ Wd + bd
        act_bf16_kernel<<<ewGrid, 256, 0, stream>>>(mbuf, xa, NF);
        gemm(xa, WdT + (long)b * F * F, bd + (long)b * F, x, nullptr, nullptr,
             x, u + (long)b * F, MODE_GATED);
        // 2 feature residual layers on x; last one also writes out[b]
        for (int r = 0; r < 2; ++r) {
            act_bf16_kernel<<<ewGrid, 256, 0, stream>>>(x, xa, NF);
            gemm(xa, Wrf1T + ((long)b * 2 + r) * F * F, brf1 + ((long)b * 2 + r) * F,
                 nullptr, ybf, nullptr, nullptr, nullptr, MODE_BIAS_ACT);
            float* o2 = (r == 1) ? (out + (long)b * NF) : nullptr;
            gemm(ybf, Wrf2T + ((long)b * 2 + r) * F * F, brf2 + ((long)b * 2 + r) * F,
                 x, nullptr, o2, x, nullptr, MODE_BIAS_RESID);
        }
    }
}
